// RSSM_33749853012759
// MI455X (gfx1250) — compile-verified, matrix operand order checked by reading, not run on hardware
//
#include <hip/hip_runtime.h>
#include <hip/hip_bf16.h>
#include <math.h>

// ---------------------------------------------------------------------------
// RSSM observe scan for MI455X (gfx1250).
//  - Sequential over T=64, parallel over B=1024 (32 WGs x 32-row batch tiles).
//  - All GEMMs via v_wmma_f32_16x16x32_bf16 (bf16 operands, f32 accum).
//  - Weights pre-swizzled once per call into WMMA B-fragment layout in d_ws
//    (bf16, ~1.23 MB -> L2 resident; re-read every step for free).
//  - GRU gates computed fully in-register (z/r/h weight slices packed
//    separately so one wave owns all six 16x16 gate tiles of a column block).
//  - Embed tile (32x1024 fp32, 128 KB) staged to LDS once per timestep with
//    GLOBAL_LOAD_ASYNC_TO_LDS_B128 (ASYNCcnt) issued at step start and waited
//    just before the obs1 GEMM -> copy overlaps steps A..E.
//  - Long K chains (img2: 7, obs1: 39) split over two accumulators for
//    tensor-pipe ILP (WMMA->WMMA RAW hazard costs ~5 co-exec slots).
// ---------------------------------------------------------------------------

typedef __attribute__((ext_vector_type(16))) __bf16 v16bf;
typedef __attribute__((ext_vector_type(8)))  float  v8f;
typedef __attribute__((ext_vector_type(4)))  int    v4i;

#if defined(__gfx1250__) && \
    __has_builtin(__builtin_amdgcn_global_load_async_to_lds_b128) && \
    __has_builtin(__builtin_amdgcn_s_wait_asynccnt)
#define USE_ASYNC 1
typedef __attribute__((address_space(1))) v4i as1_v4i;  // global int4
typedef __attribute__((address_space(3))) v4i as3_v4i;  // LDS int4
#else
#define USE_ASYNC 0
#endif

#if defined(__HIP_DEVICE_COMPILE__) && !USE_ASYNC
#warning "CDNA5 async-to-LDS builtins NOT available; using plain LDS staging"
#endif

#define ST  30
#define DD  200
#define HH  200
#define EE  1024
#define AA  6
#define BB  1024
#define TT  64

#define MT     32     // batch rows per workgroup
#define MB     2      // 16-row M blocks per tile
#define NB_H   13     // 208/16 N blocks for N=200 outputs
#define NB_S   2      // 32/16  N blocks for N=30 outputs
#define KB_I   2      // 64/32  K blocks (img1, K=36 padded to 64)
#define KB_H   7      // 224/32 K blocks (K=200 padded to 224)
#define KB_O   39     // 1248/32 K blocks (obs1: 224 deter-pad + 1024 embed)
#define KPAD_I 64
#define KPAD_H 224

// packed-weight workspace offsets (bf16 elements); each fragment = 512 elems
#define FRAG      512
#define SZ_IMG1   (KB_I*NB_H*FRAG)
#define SZ_HH     (KB_H*NB_H*FRAG)
#define SZ_S      (KB_H*NB_S*FRAG)
#define SZ_OBS1   (KB_O*NB_H*FRAG)
#define OFF_IMG1  0
#define OFF_WZ    (OFF_IMG1 + SZ_IMG1)
#define OFF_WR    (OFF_WZ   + SZ_HH)
#define OFF_WH    (OFF_WR   + SZ_HH)
#define OFF_UZ    (OFF_WH   + SZ_HH)
#define OFF_UR    (OFF_UZ   + SZ_HH)
#define OFF_UH    (OFF_UR   + SZ_HH)
#define OFF_IMG2  (OFF_UH   + SZ_HH)
#define OFF_IMEAN (OFF_IMG2 + SZ_HH)
#define OFF_ISTD  (OFF_IMEAN + SZ_S)
#define OFF_OBS1  (OFF_ISTD  + SZ_S)
#define OFF_OMEAN (OFF_OBS1  + SZ_OBS1)
#define OFF_OSTD  (OFF_OMEAN + SZ_S)

// ---------------------------------------------------------------------------
// Weight pre-swizzle: fp32 [K x N] (row-major, arbitrary column slice) ->
// bf16 fragments matching the 16-bit WMMA operand lane layout:
// lane = 16*(K-half) + column, 16 contiguous bf16 per lane per 32x16 frag.
// kMode: 0 = identity (valid k < kValid)
//        1 = img1 concat map (stoch 0..29 -> 0..29, action 30..35 -> 32..37)
//        2 = obs1 concat map (deter 0..199 -> 0..199, embed -> 224..1247)
// ---------------------------------------------------------------------------
__global__ void __launch_bounds__(256) pack_w_kernel(
    const float* __restrict__ src, int srcLd, int nOff,
    int kMode, int kValid, int nValid,
    int KB, int NBc, __bf16* __restrict__ dst)
{
  int tid = blockIdx.x * blockDim.x + threadIdx.x;
  int nf = KB * NBc;
  if (tid >= nf * 32) return;
  int frag = tid >> 5, lane = tid & 31;
  int kb = frag / NBc, nb = frag - kb * NBc;
  int n = nb * 16 + (lane & 15);
  int kofs = kb * 32 + ((lane >> 4) << 3);
  v16bf out;
#pragma unroll
  for (int i = 0; i < 16; ++i) {
    int v = i >> 1, h = i & 1;
    int k = kofs + ((v < 4) ? v * 2 : 16 + (v - 4) * 2) + h;
    int ks;
    if (kMode == 1)      ks = (k < 30) ? k : ((k >= 32 && k < 38) ? k - 2 : -1);
    else if (kMode == 2) ks = (k < 200) ? k : ((k >= 224 && k < 1248) ? k - 24 : -1);
    else                 ks = (k < kValid) ? k : -1;
    float val = 0.f;
    if (ks >= 0 && n < nValid) val = src[(size_t)ks * srcLd + nOff + n];
    out[i] = (__bf16)val;
  }
  *(v16bf*)(dst + ((size_t)frag * 32 + lane) * 16) = out;
}

// ---------------------------------------------------------------------------
// Fragment helpers
// ---------------------------------------------------------------------------
__device__ __forceinline__ v8f wmma_bf(v16bf a, v16bf b, v8f c) {
  return __builtin_amdgcn_wmma_f32_16x16x32_bf16(
      false, a, false, b, (short)0, c, false, false);
}

// A fragment (16x32 bf16) gathered from an LDS tile stored row-major
// (elements 0..7 -> k 0..7, 8..15 -> k 16..23 => two ds_load_b128 per lane)
__device__ __forceinline__ v16bf afrag_lds(const __bf16* base, int stride,
                                           int kb, int mb, int lane) {
  const __bf16* p = base + (mb * 16 + (lane & 15)) * stride
                         + kb * 32 + ((lane >> 4) << 3);
  v16bf f;
#pragma unroll
  for (int i = 0; i < 16; ++i) {
    int v = i >> 1, h = i & 1;
    f[i] = p[((v < 4) ? v * 2 : 16 + (v - 4) * 2) + h];
  }
  return f;
}

// A fragment gathered from an fp32 LDS tile (embed), convert to bf16
__device__ __forceinline__ v16bf afrag_ldsf32(const float* base, int stride,
                                              int kb, int mb, int lane) {
  const float* p = base + (mb * 16 + (lane & 15)) * stride
                        + kb * 32 + ((lane >> 4) << 3);
  v16bf f;
#pragma unroll
  for (int i = 0; i < 16; ++i) {
    int v = i >> 1, h = i & 1;
    f[i] = (__bf16)p[((v < 4) ? v * 2 : 16 + (v - 4) * 2) + h];
  }
  return f;
}

// Pre-swizzled B fragment: one aligned 32-byte load per lane
__device__ __forceinline__ v16bf bfrag(const __bf16* pk, int kb, int NBc,
                                       int nb, int lane) {
  return *(const v16bf*)(pk + ((size_t)(kb * NBc + nb) * 32 + lane) * 16);
}

__device__ __forceinline__ float elu_f(float v) {
  return v > 0.f ? v : (__expf(v) - 1.f);
}
__device__ __forceinline__ float sigmoid_f(float v) {
  return 1.f / (1.f + __expf(-v));
}
__device__ __forceinline__ float softplus_f(float v) {
  return v > 20.f ? v : log1pf(__expf(v));
}

// ---------------------------------------------------------------------------
// Main RSSM kernel: one workgroup owns MT=32 batch rows, loops t=0..63.
// ---------------------------------------------------------------------------
__global__ void __launch_bounds__(256) rssm_kernel(
    const float* __restrict__ embed,   // [B,T,E]
    const float* __restrict__ action,  // [B,T,A]
    const float* __restrict__ nzp,     // [T,B,S]
    const float* __restrict__ nzo,     // [T,B,S]
    const float* __restrict__ b_img1, const float* __restrict__ gru_b,
    const float* __restrict__ gru_rb, const float* __restrict__ b_img2,
    const float* __restrict__ b_imean, const float* __restrict__ b_istd,
    const float* __restrict__ b_obs1, const float* __restrict__ b_omean,
    const float* __restrict__ b_ostd,
    const __bf16* __restrict__ wpk,    // packed weights (workspace)
    float* __restrict__ out)           // [B,T,580]
{
  extern __shared__ char smem[];
  float*  deterF = (float*)smem;                         // MT*DD fp32 state
  __bf16* db0 = (__bf16*)(smem + MT * DD * 4);           // deter bf16 (ping)
  __bf16* db1 = db0 + MT * KPAD_H;                       // deter bf16 (pong)
  __bf16* xbuf = db1 + MT * KPAD_H;                      // activation A tile
  __bf16* in1 = xbuf + MT * KPAD_H;                      // [stoch|action] tile
  float*  ebuf = (float*)(in1 + MT * KPAD_I);            // MT*EE fp32 embed

  const int tid = threadIdx.x, lane = tid & 31, wave = tid >> 5;
  const int b0 = blockIdx.x * MT;
  const int hi = lane >> 4;

  const __bf16* Pimg1 = wpk + OFF_IMG1;
  const __bf16* Pwz = wpk + OFF_WZ;  const __bf16* Pwr = wpk + OFF_WR;
  const __bf16* Pwh = wpk + OFF_WH;  const __bf16* Puz = wpk + OFF_UZ;
  const __bf16* Pur = wpk + OFF_UR;  const __bf16* Puh = wpk + OFF_UH;
  const __bf16* Pimg2 = wpk + OFF_IMG2;
  const __bf16* Pimean = wpk + OFF_IMEAN; const __bf16* Pistd = wpk + OFF_ISTD;
  const __bf16* Pobs1 = wpk + OFF_OBS1;
  const __bf16* Pomean = wpk + OFF_OMEAN; const __bf16* Postd = wpk + OFF_OSTD;

  // init: deter=0, stoch=0, all pads zero
  for (int i = tid; i < MT * DD; i += 256) deterF[i] = 0.f;
  for (int i = tid; i < MT * KPAD_H; i += 256) {
    db0[i] = (__bf16)0.f; db1[i] = (__bf16)0.f; xbuf[i] = (__bf16)0.f;
  }
  for (int i = tid; i < MT * KPAD_I; i += 256) in1[i] = (__bf16)0.f;
  __syncthreads();

  for (int t = 0; t < TT; ++t) {
    const __bf16* Dold = (t & 1) ? db1 : db0;
    __bf16* Dnew = (t & 1) ? db0 : db1;

    // ---- kick off embed tile stage (fp32, 128 KB) for step F; overlaps A..E
    {
      // MT*EE/4 = 8192 16-byte chunks; 256 threads x 32 chunks
      for (int i = tid; i < MT * EE / 4; i += 256) {
        int m = i >> 8;            // 256 chunks per row
        int q = (i & 255) << 2;    // float offset within row
        const float* gp = embed + ((size_t)(b0 + m) * TT + t) * EE + q;
        float* lp = ebuf + m * EE + q;
#if USE_ASYNC
        __builtin_amdgcn_global_load_async_to_lds_b128(
            (as1_v4i*)(size_t)gp, (as3_v4i*)lp, 0, 0);
#else
        *(float4*)lp = *(const float4*)gp;
#endif
      }
    }

    // ---- stage action into in1 cols 32..37 (stoch cols 0..29 set by step G)
    for (int i = tid; i < MT * AA; i += 256) {
      int m = i / AA, j = i - m * AA;
      in1[m * KPAD_I + 32 + j] =
          (__bf16)action[((size_t)(b0 + m) * TT + t) * AA + j];
    }
    __syncthreads();

    // ---- B: x = elu([stoch,a] @ w_img1 + b_img1) -> xbuf
    for (int task = wave; task < MB * NB_H; task += 8) {
      int mb = task & 1, nb = task >> 1;
      v8f acc = {};
      for (int kb = 0; kb < KB_I; ++kb)
        acc = wmma_bf(afrag_lds(in1, KPAD_I, kb, mb, lane),
                      bfrag(Pimg1, kb, NB_H, nb, lane), acc);
      int c = nb * 16 + (lane & 15);
      if (c < HH) {
        float bv = b_img1[c];
#pragma unroll
        for (int p = 0; p < 8; ++p)
          xbuf[(mb * 16 + p + hi * 8) * KPAD_H + c] = (__bf16)elu_f(acc[p] + bv);
      }
    }
    __syncthreads();

    // ---- C: GRU gates, fully in-register per 16x16 column tile
    for (int task = wave; task < MB * NB_H; task += 8) {
      int mb = task & 1, nb = task >> 1;
      v8f xz = {}, xr = {}, xh = {}, hz = {}, hr = {}, hh = {};
      for (int kb = 0; kb < KB_H; ++kb) {
        v16bf ax = afrag_lds(xbuf, KPAD_H, kb, mb, lane);
        v16bf ad = afrag_lds(Dold, KPAD_H, kb, mb, lane);
        xz = wmma_bf(ax, bfrag(Pwz, kb, NB_H, nb, lane), xz);
        xr = wmma_bf(ax, bfrag(Pwr, kb, NB_H, nb, lane), xr);
        xh = wmma_bf(ax, bfrag(Pwh, kb, NB_H, nb, lane), xh);
        hz = wmma_bf(ad, bfrag(Puz, kb, NB_H, nb, lane), hz);
        hr = wmma_bf(ad, bfrag(Pur, kb, NB_H, nb, lane), hr);
        hh = wmma_bf(ad, bfrag(Puh, kb, NB_H, nb, lane), hh);
      }
      int c = nb * 16 + (lane & 15);
      if (c < DD) {
        float bz = gru_b[c] + gru_rb[c];
        float br = gru_b[DD + c] + gru_rb[DD + c];
        float bxh = gru_b[2 * DD + c], bhh = gru_rb[2 * DD + c];
#pragma unroll
        for (int p = 0; p < 8; ++p) {
          int m = mb * 16 + p + hi * 8;
          float z = sigmoid_f(xz[p] + hz[p] + bz);
          float r = sigmoid_f(xr[p] + hr[p] + br);
          float hv = tanhf(xh[p] + bxh + r * (hh[p] + bhh));
          float dn = z * deterF[m * DD + c] + (1.f - z) * hv;
          deterF[m * DD + c] = dn;
          Dnew[m * KPAD_H + c] = (__bf16)dn;
        }
      }
    }
    __syncthreads();

    // ---- deter output copies (posterior @90, prior @380)
    for (int i = tid; i < MT * DD; i += 256) {
      int m = i / DD, c = i - m * DD;
      size_t ob = ((size_t)(b0 + m) * TT + t) * 580;
      float dv = deterF[i];
      out[ob + 90 + c] = dv;
      out[ob + 380 + c] = dv;
    }

    // ---- D: x2 = elu(deter_n @ w_img2 + b_img2) -> xbuf (2 accumulators)
    for (int task = wave; task < MB * NB_H; task += 8) {
      int mb = task & 1, nb = task >> 1;
      v8f a0 = {}, a1 = {};
      for (int kb = 0; kb + 1 < KB_H; kb += 2) {
        a0 = wmma_bf(afrag_lds(Dnew, KPAD_H, kb, mb, lane),
                     bfrag(Pimg2, kb, NB_H, nb, lane), a0);
        a1 = wmma_bf(afrag_lds(Dnew, KPAD_H, kb + 1, mb, lane),
                     bfrag(Pimg2, kb + 1, NB_H, nb, lane), a1);
      }
      a0 = wmma_bf(afrag_lds(Dnew, KPAD_H, KB_H - 1, mb, lane),
                   bfrag(Pimg2, KB_H - 1, NB_H, nb, lane), a0);
      v8f acc = a0 + a1;
      int c = nb * 16 + (lane & 15);
      if (c < HH) {
        float bv = b_img2[c];
#pragma unroll
        for (int p = 0; p < 8; ++p)
          xbuf[(mb * 16 + p + hi * 8) * KPAD_H + c] = (__bf16)elu_f(acc[p] + bv);
      }
    }
    __syncthreads();

    // ---- E: prior stats (pmean, pstd, pstoch)
    if (wave < MB * NB_S) {
      int mb = wave & 1, nb = wave >> 1;
      v8f am = {}, as = {};
      for (int kb = 0; kb < KB_H; ++kb) {
        v16bf a = afrag_lds(xbuf, KPAD_H, kb, mb, lane);
        am = wmma_bf(a, bfrag(Pimean, kb, NB_S, nb, lane), am);
        as = wmma_bf(a, bfrag(Pistd, kb, NB_S, nb, lane), as);
      }
      int c = nb * 16 + (lane & 15);
      if (c < ST) {
        float bm = b_imean[c], bs = b_istd[c];
#pragma unroll
        for (int p = 0; p < 8; ++p) {
          int b = b0 + mb * 16 + p + hi * 8;
          float mean = am[p] + bm;
          float sd = softplus_f(as[p] + bs) + 0.1f;
          float stv = mean + sd * nzp[((size_t)t * BB + b) * ST + c];
          size_t ob = ((size_t)b * TT + t) * 580;
          out[ob + 290 + c] = mean;
          out[ob + 320 + c] = sd;
          out[ob + 350 + c] = stv;
        }
      }
    }
#if USE_ASYNC
    __builtin_amdgcn_s_wait_asynccnt(0);   // embed tile resident in LDS
#endif
    __syncthreads();

    // ---- F: xo = elu([deter_n, embed] @ w_obs1 + b_obs1) -> xbuf
    //      (2 accumulators over the 39-deep K chain)
    for (int task = wave; task < MB * NB_H; task += 8) {
      int mb = task & 1, nb = task >> 1;
      v8f a0 = {}, a1 = {};
      for (int kb = 0; kb + 1 < KB_H; kb += 2) {
        a0 = wmma_bf(afrag_lds(Dnew, KPAD_H, kb, mb, lane),
                     bfrag(Pobs1, kb, NB_H, nb, lane), a0);
        a1 = wmma_bf(afrag_lds(Dnew, KPAD_H, kb + 1, mb, lane),
                     bfrag(Pobs1, kb + 1, NB_H, nb, lane), a1);
      }
      a0 = wmma_bf(afrag_lds(Dnew, KPAD_H, KB_H - 1, mb, lane),
                   bfrag(Pobs1, KB_H - 1, NB_H, nb, lane), a0);
      for (int kb = KB_H; kb + 1 < KB_O; kb += 2) {   // 32 embed blocks, even
        a0 = wmma_bf(afrag_ldsf32(ebuf, EE, kb - KB_H, mb, lane),
                     bfrag(Pobs1, kb, NB_H, nb, lane), a0);
        a1 = wmma_bf(afrag_ldsf32(ebuf, EE, kb + 1 - KB_H, mb, lane),
                     bfrag(Pobs1, kb + 1, NB_H, nb, lane), a1);
      }
      v8f acc = a0 + a1;
      int c = nb * 16 + (lane & 15);
      if (c < HH) {
        float bv = b_obs1[c];
#pragma unroll
        for (int p = 0; p < 8; ++p)
          xbuf[(mb * 16 + p + hi * 8) * KPAD_H + c] = (__bf16)elu_f(acc[p] + bv);
      }
    }
    __syncthreads();

    // ---- G: posterior stats (omean, ostd, ostoch) + stoch feedback into in1
    if (wave < MB * NB_S) {
      int mb = wave & 1, nb = wave >> 1;
      v8f am = {}, as = {};
      for (int kb = 0; kb < KB_H; ++kb) {
        v16bf a = afrag_lds(xbuf, KPAD_H, kb, mb, lane);
        am = wmma_bf(a, bfrag(Pomean, kb, NB_S, nb, lane), am);
        as = wmma_bf(a, bfrag(Postd, kb, NB_S, nb, lane), as);
      }
      int c = nb * 16 + (lane & 15);
      if (c < ST) {
        float bm = b_omean[c], bs = b_ostd[c];
#pragma unroll
        for (int p = 0; p < 8; ++p) {
          int m = mb * 16 + p + hi * 8;
          int b = b0 + m;
          float mean = am[p] + bm;
          float sd = softplus_f(as[p] + bs);
          float stv = mean + sd * nzo[((size_t)t * BB + b) * ST + c];
          size_t ob = ((size_t)b * TT + t) * 580;
          out[ob + 0 + c] = mean;
          out[ob + 30 + c] = sd;
          out[ob + 60 + c] = stv;
          in1[m * KPAD_I + c] = (__bf16)stv;
        }
      }
    }
    __syncthreads();
  }
}

// ---------------------------------------------------------------------------
extern "C" void kernel_launch(void* const* d_in, const int* in_sizes, int n_in,
                              void* d_out, int out_size, void* d_ws,
                              size_t ws_size, hipStream_t stream) {
  const float* embed   = (const float*)d_in[0];
  const float* action  = (const float*)d_in[1];
  const float* nzp     = (const float*)d_in[2];
  const float* nzo     = (const float*)d_in[3];
  const float* w_img1  = (const float*)d_in[4];
  const float* b_img1  = (const float*)d_in[5];
  const float* gru_w   = (const float*)d_in[6];
  const float* gru_u   = (const float*)d_in[7];
  const float* gru_b   = (const float*)d_in[8];
  const float* gru_rb  = (const float*)d_in[9];
  const float* w_img2  = (const float*)d_in[10];
  const float* b_img2  = (const float*)d_in[11];
  const float* w_imean = (const float*)d_in[12];
  const float* b_imean = (const float*)d_in[13];
  const float* w_istd  = (const float*)d_in[14];
  const float* b_istd  = (const float*)d_in[15];
  const float* w_obs1  = (const float*)d_in[16];
  const float* b_obs1  = (const float*)d_in[17];
  const float* w_omean = (const float*)d_in[18];
  const float* b_omean = (const float*)d_in[19];
  const float* w_ostd  = (const float*)d_in[20];
  const float* b_ostd  = (const float*)d_in[21];
  __bf16* wpk = (__bf16*)d_ws;

  auto pack = [&](const float* src, int ld, int nOff, int kMode, int kValid,
                  int nValid, int KB, int NBc, size_t off) {
    int threads = KB * NBc * 32;
    int blocks = (threads + 255) / 256;
    pack_w_kernel<<<blocks, 256, 0, stream>>>(src, ld, nOff, kMode, kValid,
                                              nValid, KB, NBc, wpk + off);
  };

  pack(w_img1, 200, 0, 1, 0, 200, KB_I, NB_H, OFF_IMG1);
  pack(gru_w, 600, 0,   0, 200, 200, KB_H, NB_H, OFF_WZ);
  pack(gru_w, 600, 200, 0, 200, 200, KB_H, NB_H, OFF_WR);
  pack(gru_w, 600, 400, 0, 200, 200, KB_H, NB_H, OFF_WH);
  pack(gru_u, 600, 0,   0, 200, 200, KB_H, NB_H, OFF_UZ);
  pack(gru_u, 600, 200, 0, 200, 200, KB_H, NB_H, OFF_UR);
  pack(gru_u, 600, 400, 0, 200, 200, KB_H, NB_H, OFF_UH);
  pack(w_img2, 200, 0, 0, 200, 200, KB_H, NB_H, OFF_IMG2);
  pack(w_imean, 30, 0, 0, 200, 30, KB_H, NB_S, OFF_IMEAN);
  pack(w_istd,  30, 0, 0, 200, 30, KB_H, NB_S, OFF_ISTD);
  pack(w_obs1, 200, 0, 2, 0, 200, KB_O, NB_H, OFF_OBS1);
  pack(w_omean, 30, 0, 0, 200, 30, KB_H, NB_S, OFF_OMEAN);
  pack(w_ostd,  30, 0, 0, 200, 30, KB_H, NB_S, OFF_OSTD);

  size_t shm = (size_t)MT * DD * 4            // deterF fp32
             + (size_t)MT * KPAD_H * 2 * 3    // db0, db1, xbuf bf16
             + (size_t)MT * KPAD_I * 2        // in1 bf16
             + (size_t)MT * EE * 4;           // ebuf fp32 (embed tile)
  rssm_kernel<<<BB / MT, 256, shm, stream>>>(
      embed, action, nzp, nzo, b_img1, gru_b, gru_rb, b_img2, b_imean, b_istd,
      b_obs1, b_omean, b_ostd, wpk, (float*)d_out);
}
